// LSTM1_10849087390020
// MI455X (gfx1250) — matrix operand order repaired and checked by reading, not compiled, not run on hardware
//
#include <hip/hip_runtime.h>
#include <hip/hip_fp16.h>
#include <math.h>

typedef _Float16 half_t;
typedef __attribute__((ext_vector_type(16))) _Float16 v16h;
typedef __attribute__((ext_vector_type(8)))  float    v8f;

#define B_  256
#define S_  512
#define I_  512
#define H_  1024
#define K_  1536      // I_ + H_
#define KT_ 48        // K_/32 k-tiles (16 from x, 32 from h)
#define NG_ 4096      // 4 gates * H_

// ---------------------------------------------------------------------------
// Prepare: convert all gate weights f32->f16 and pre-swizzle into WMMA
// B-fragment order.  Concatenated weight Wcat is (K=1536) x (N=4096):
//   rows 0..511    = X_{i,f,g,o}[k][h]
//   rows 512..1535 = H_{i,f,g,o}[k-512][h]
//   cols: gate-major, n = gate*1024 + h, gate order i,f,g,o.
// B fragment order used here (contiguous k, same as A; 32x16 tile):
//   lane = nn | ((kk>>4)<<4), j = kk & 15  -> each lane holds 16 contig halfs.
// Fragment-ready address: ((nt*KT + kt)*32 + lane)*16 + j
// ---------------------------------------------------------------------------
__global__ __launch_bounds__(256)
void prep_weights(const float* __restrict__ Xi, const float* __restrict__ Hi,
                  const float* __restrict__ Xf, const float* __restrict__ Hf,
                  const float* __restrict__ Xg, const float* __restrict__ Hg,
                  const float* __restrict__ Xo, const float* __restrict__ Ho,
                  half_t* __restrict__ wcat) {
    long id = (long)blockIdx.x * blockDim.x + threadIdx.x;
    if (id >= (long)K_ * NG_) return;
    int k    = (int)(id >> 12);     // / 4096
    int n    = (int)(id & 4095);
    int gate = n >> 10;
    int h    = n & 1023;
    float v;
    if (k < I_) {
        const float* xs[4] = {Xi, Xf, Xg, Xo};
        v = xs[gate][(long)k * H_ + h];
    } else {
        const float* hs[4] = {Hi, Hf, Hg, Ho};
        v = hs[gate][(long)(k - I_) * H_ + h];
    }
    int kt = k >> 5, kk = k & 31;
    int nt = n >> 4, nn = n & 15;
    int lane = nn | ((kk >> 4) << 4);
    int j    = kk & 15;
    long dst = ((long)(nt * KT_ + kt) * 32 + lane) * 16 + j;
    wcat[dst] = (half_t)v;
}

// init h (f16 ping buffer) + c (f32) + concatenated bias
__global__ __launch_bounds__(256)
void prep_state(const float* __restrict__ h0, const float* __restrict__ c0,
                const float* __restrict__ bi, const float* __restrict__ bf,
                const float* __restrict__ bg, const float* __restrict__ bo,
                half_t* __restrict__ h16_0, float* __restrict__ c,
                float* __restrict__ bcat) {
    int id = blockIdx.x * blockDim.x + threadIdx.x;
    if (id < B_ * H_) {
        h16_0[id] = (half_t)h0[id];
        c[id]     = c0[id];
    }
    if (id < NG_) {
        const float* bs[4] = {bi, bf, bg, bo};
        bcat[id] = bs[id >> 10][id & 1023];
    }
}

__device__ __forceinline__ float sigmoid_f(float x) {
    return 1.0f / (1.0f + __expf(-x));
}

__device__ __forceinline__ unsigned pack_h2(float a, float b) {
    half_t ha = (half_t)a, hb = (half_t)b;
    unsigned short ua = __builtin_bit_cast(unsigned short, ha);
    unsigned short ub = __builtin_bit_cast(unsigned short, hb);
    return (unsigned)ua | ((unsigned)ub << 16);
}

// ---------------------------------------------------------------------------
// One LSTM timestep, fully fused:
//   gates = [x_t | h_{t-1}] (256x1536 f16)  @  Wcat (1536x4096 f16) + bias
//   c' = sigm(f)*c + sigm(i)*tanh(g);  h' = sigm(o)*tanh(c')
// Grid: (M/16 = 16) x (H/64 = 16).  Block: 256 threads = 8 waves.
// Each block owns a 16x64 slice of all four gates (16 WMMA tiles; 2 per wave).
// A tile double-buffered through LDS in fragment order (1 barrier / k-iter);
// B fragments stream pre-swizzled from L2 (contiguous 32B per lane).
// K loop is split: first 16 tiles read x (f32->f16), last 32 read h (f16).
// ---------------------------------------------------------------------------
__global__ __launch_bounds__(256)
void lstm_step(const float*  __restrict__ x,      // (B,S,I) f32
               const half_t* __restrict__ wcat,   // swizzled (K x 4H) f16
               const float*  __restrict__ bcat,   // (4H) f32
               const half_t* __restrict__ h_prev, // (B,H) f16
               half_t*       __restrict__ h_next, // (B,H) f16
               float*        __restrict__ h_f32,  // (B,H) f32
               float*        __restrict__ c,      // (B,H) f32
               int s) {
    __shared__ __align__(32) half_t ldsA[2][16 * 32];    // double-buffered A
    __shared__ __align__(32) float  ldsG[4 * 16 * 64];   // gate pre-activations

    const int tid  = threadIdx.x;
    const int lane = tid & 31;
    const int wave = tid >> 5;
    const int m0   = blockIdx.x * 16;    // batch-row base
    const int ntb  = blockIdx.y;         // which 64-wide h slice

    // wave -> two 16x16 tiles: t = gate*4 + sub
    const int t0 = wave * 2, t1 = t0 + 1;
    const int g0 = t0 >> 2, sub0 = t0 & 3;
    const int g1 = t1 >> 2, sub1 = t1 & 3;
    const int NT0 = g0 * 64 + ntb * 4 + sub0;   // global 16-wide n-tile index
    const int NT1 = g1 * 64 + ntb * 4 + sub1;

    // staging role of this thread: one (row m, even K pair kk) per k-tile
    const int sm  = tid >> 4;            // 0..15 row
    const int skk = (tid & 15) * 2;      // 0,2,..,30 (even)
    // fragment mapping: lane = m | ((kk>>4)<<4); j = kk&15
    const int sln = sm | ((skk >> 4) << 4);
    const int sj  = skk & 15;     // even -> dword slot
    unsigned* const dstw0 = (unsigned*)&ldsA[0][sln * 16 + sj];
    unsigned* const dstw1 = (unsigned*)&ldsA[1][sln * 16 + sj];

    // streaming source pointers (advance 32 elements per k-tile)
    const float*  xsrc = x + ((long)(m0 + sm) * S_ + s) * I_ + skk;
    const half_t* hsrc = h_prev + (long)(m0 + sm) * H_ + skk;

    // pre-swizzled B streams (advance 512 halfs = one 32x16 tile per k-tile)
    const half_t* wb0 = wcat + (long)NT0 * (KT_ * 512) + lane * 16;
    const half_t* wb1 = wcat + (long)NT1 * (KT_ * 512) + lane * 16;

    v8f acc0 = {};
    v8f acc1 = {};
    int p = 0;

    // ---- phase 1: K from x (16 k-tiles) ----
    for (int kt = 0; kt < 16; ++kt) {
        float2 v2 = *(const float2*)xsrc;  xsrc += 32;
        *(p ? dstw1 : dstw0) = pack_h2(v2.x, v2.y);
        __syncthreads();
        v16h afrag = *(const v16h*)(&ldsA[p][lane * 16]);
        v16h b0 = *(const v16h*)wb0;  wb0 += 512;
        v16h b1 = *(const v16h*)wb1;  wb1 += 512;
        acc0 = __builtin_amdgcn_wmma_f32_16x16x32_f16(
                   false, afrag, false, b0, (short)0, acc0, false, false);
        acc1 = __builtin_amdgcn_wmma_f32_16x16x32_f16(
                   false, afrag, false, b1, (short)0, acc1, false, false);
        p ^= 1;
    }

    // ---- phase 2: K from h_prev (32 k-tiles), already f16: copy dwords ----
    for (int kt = 0; kt < 32; ++kt) {
        unsigned hw = *(const unsigned*)hsrc;  hsrc += 32;
        *(p ? dstw1 : dstw0) = hw;
        __syncthreads();
        v16h afrag = *(const v16h*)(&ldsA[p][lane * 16]);
        v16h b0 = *(const v16h*)wb0;  wb0 += 512;
        v16h b1 = *(const v16h*)wb1;  wb1 += 512;
        acc0 = __builtin_amdgcn_wmma_f32_16x16x32_f16(
                   false, afrag, false, b0, (short)0, acc0, false, false);
        acc1 = __builtin_amdgcn_wmma_f32_16x16x32_f16(
                   false, afrag, false, b1, (short)0, acc1, false, false);
        p ^= 1;
    }

    // ---- scatter accumulators to ldsG: D layout lane->N=lane&15,
    //      VGPR r -> M = r + 8*(lane>=16) ----
    {
        int N   = lane & 15;
        int mhi = (lane >> 4) << 3;
        #pragma unroll
        for (int r = 0; r < 8; ++r) {
            int M = r + mhi;
            ldsG[g0 * 1024 + M * 64 + sub0 * 16 + N] = acc0[r];
            ldsG[g1 * 1024 + M * 64 + sub1 * 16 + N] = acc1[r];
        }
    }
    __syncthreads();

    // ---- pointwise LSTM update; each thread owns 4 of the 16x64 elements ----
    #pragma unroll
    for (int q = 0; q < 4; ++q) {
        int e   = tid * 4 + q;      // 0..1023
        int m   = e >> 6;
        int col = e & 63;
        int hidx = ntb * 64 + col;
        float pi = ldsG[0 * 1024 + m * 64 + col] + bcat[0 * H_ + hidx];
        float pf = ldsG[1 * 1024 + m * 64 + col] + bcat[1 * H_ + hidx];
        float pg = ldsG[2 * 1024 + m * 64 + col] + bcat[2 * H_ + hidx];
        float po = ldsG[3 * 1024 + m * 64 + col] + bcat[3 * H_ + hidx];
        float it = sigmoid_f(pi);
        float ft = sigmoid_f(pf);
        float gt = tanhf(pg);
        float ot = sigmoid_f(po);
        long idx = (long)(m0 + m) * H_ + hidx;
        float cn = ft * c[idx] + it * gt;
        float hn = ot * tanhf(cn);
        c[idx]      = cn;
        h_f32[idx]  = hn;
        h_next[idx] = (half_t)hn;
    }
}

// ---------------------------------------------------------------------------
// Per-step head: out[b, s, :] = softmax(h[b,:] @ W_out + b_out)   (OUT = 3)
// ---------------------------------------------------------------------------
__global__ __launch_bounds__(128)
void proj_softmax(const float* __restrict__ h_f32,
                  const float* __restrict__ Wout,   // (H,3)
                  const float* __restrict__ bout,   // (3,)
                  float* __restrict__ out,          // (B,S,3)
                  int s) {
    __shared__ float red[128 * 3];
    int b   = blockIdx.x;
    int tid = threadIdx.x;
    float p0 = 0.f, p1 = 0.f, p2 = 0.f;
    for (int k = tid; k < H_; k += 128) {
        float hv = h_f32[(long)b * H_ + k];
        p0 += hv * Wout[k * 3 + 0];
        p1 += hv * Wout[k * 3 + 1];
        p2 += hv * Wout[k * 3 + 2];
    }
    red[tid] = p0; red[128 + tid] = p1; red[256 + tid] = p2;
    __syncthreads();
    for (int st = 64; st > 0; st >>= 1) {
        if (tid < st) {
            red[tid]       += red[tid + st];
            red[128 + tid] += red[128 + tid + st];
            red[256 + tid] += red[256 + tid + st];
        }
        __syncthreads();
    }
    if (tid == 0) {
        float l0 = red[0]   + bout[0];
        float l1 = red[128] + bout[1];
        float l2 = red[256] + bout[2];
        float mx = fmaxf(l0, fmaxf(l1, l2));
        float e0 = __expf(l0 - mx), e1 = __expf(l1 - mx), e2 = __expf(l2 - mx);
        float inv = 1.0f / (e0 + e1 + e2);
        float* o = out + ((long)b * S_ + s) * 3;
        o[0] = e0 * inv; o[1] = e1 * inv; o[2] = e2 * inv;
    }
}

// copy final (h, c) into the output tuple tail
__global__ __launch_bounds__(256)
void finalize(const float* __restrict__ h_f32, const float* __restrict__ c,
              float* __restrict__ out) {
    int id = blockIdx.x * blockDim.x + threadIdx.x;
    if (id < B_ * H_) {
        out[B_ * S_ * 3 + id]           = h_f32[id];
        out[B_ * S_ * 3 + B_ * H_ + id] = c[id];
    }
}

// ---------------------------------------------------------------------------
extern "C" void kernel_launch(void* const* d_in, const int* in_sizes, int n_in,
                              void* d_out, int out_size, void* d_ws, size_t ws_size,
                              hipStream_t stream) {
    const float* x    = (const float*)d_in[0];
    const float* h0   = (const float*)d_in[1];
    const float* c0   = (const float*)d_in[2];
    const float* Xi   = (const float*)d_in[3];
    const float* Hi   = (const float*)d_in[4];
    const float* bi   = (const float*)d_in[5];
    const float* Xf   = (const float*)d_in[6];
    const float* Hf   = (const float*)d_in[7];
    const float* bf   = (const float*)d_in[8];
    const float* Xg   = (const float*)d_in[9];
    const float* Hg   = (const float*)d_in[10];
    const float* bg   = (const float*)d_in[11];
    const float* Xo   = (const float*)d_in[12];
    const float* Ho   = (const float*)d_in[13];
    const float* bo   = (const float*)d_in[14];
    const float* Wout = (const float*)d_in[15];
    const float* bout = (const float*)d_in[16];
    float* out = (float*)d_out;

    // workspace layout (~15.7 MB total)
    char* ws = (char*)d_ws;
    half_t* wcat = (half_t*)ws;                              // 12,582,912 B
    float*  bcat = (float*)(ws + 12582912);                  //     16,384 B
    half_t* h16a = (half_t*)(ws + 12599296);                 //    524,288 B
    half_t* h16b = (half_t*)(ws + 12599296 + 524288);        //    524,288 B
    float*  hf32 = (float*)(ws + 13647872);                  //  1,048,576 B
    float*  cbuf = (float*)(ws + 14696448);                  //  1,048,576 B

    {
        long total  = (long)K_ * NG_;
        int  blocks = (int)((total + 255) / 256);
        prep_weights<<<blocks, 256, 0, stream>>>(Xi, Hi, Xf, Hf, Xg, Hg, Xo, Ho, wcat);
    }
    prep_state<<<(B_ * H_ + 255) / 256, 256, 0, stream>>>(
        h0, c0, bi, bf, bg, bo, h16a, cbuf, bcat);

    half_t* hp = h16a;
    half_t* hn = h16b;
    dim3 grid(B_ / 16, H_ / 64);   // 16 x 16 blocks
    for (int s = 0; s < S_; ++s) {
        lstm_step<<<grid, 256, 0, stream>>>(x, wcat, bcat, hp, hn, hf32, cbuf, s);
        proj_softmax<<<B_, 128, 0, stream>>>(hf32, Wout, bout, out, s);
        half_t* tmp = hp; hp = hn; hn = tmp;
    }
    finalize<<<(B_ * H_ + 255) / 256, 256, 0, stream>>>(hf32, cbuf, out);
}
